// Readout_52012053954614
// MI455X (gfx1250) — matrix operand order repaired and checked by reading, not compiled, not run on hardware
//
#include <hip/hip_runtime.h>
#include <hip/hip_bf16.h>
#include <cstdint>

#define N_NODES 8192
#define D_FEAT  256
#define OUT_DIM 256
#define KTILE   32
#define KPAD    40   // padded LDS row stride (bf16 elems): 80 B = 20 dwords -> conflict-free

typedef __bf16 v16bf __attribute__((ext_vector_type(16)));
typedef float  v8f   __attribute__((ext_vector_type(8)));
typedef float  f32x4 __attribute__((ext_vector_type(4)));
typedef unsigned int u32x4 __attribute__((ext_vector_type(4)));

union Frag {
    struct { u32x4 lo, hi; } u;
    v16bf v;
};

__device__ __forceinline__ unsigned short f2bf(float f) {
    unsigned int u = __float_as_uint(f);
    u += 0x7FFFu + ((u >> 16) & 1u);   // round-to-nearest-even
    return (unsigned short)(u >> 16);
}
__device__ __forceinline__ unsigned int pack2(float a, float b) {
    return (unsigned int)f2bf(a) | ((unsigned int)f2bf(b) << 16);
}
__device__ __forceinline__ float bf2f(unsigned short h) {
    return __uint_as_float(((unsigned int)h) << 16);
}

// Load A fragment: lane (m = lane&15) holds K[8h..8h+7] and K[16+8h..16+8h+7]
__device__ __forceinline__ void load_a(Frag& a, const unsigned short* lds,
                                       int mrow, int nlo, int half) {
    const unsigned short* ap = &lds[(mrow + nlo) * KPAD + 8 * half];
    a.u.lo = *reinterpret_cast<const u32x4*>(ap);
    a.u.hi = *reinterpret_cast<const u32x4*>(ap + 16);
}
// Load B fragment: lane (n = lane&15) holds K[8h..8h+7] and K[16+8h..16+8h+7] of column crow
__device__ __forceinline__ void load_b(Frag& b, const unsigned short* lds,
                                       int crow, int nlo, int half) {
    const unsigned short* bp = &lds[(crow + nlo) * KPAD + 8 * half];
    b.u.lo = *reinterpret_cast<const u32x4*>(bp);
    b.u.hi = *reinterpret_cast<const u32x4*>(bp + 16);
}

// ---------------------------------------------------------------------------
// Kernel 1: agg = adj @ node  (bf16 WMMA, f32 acc) fused with
//           diag[i] = sum_j adj[i,j] * edge[j,i]
// 128 blocks x 256 threads; block owns 64 rows. 8 waves = 2 m-pair groups x 4
// column quarters; each wave: 2 m-tiles x 4 n-tiles (B frag reused 2x).
// ---------------------------------------------------------------------------
__global__ __launch_bounds__(256) void agg_diag_kernel(
    const float* __restrict__ adj, const float* __restrict__ node,
    const float* __restrict__ edge, float* __restrict__ agg,
    float* __restrict__ diag)
{
    __shared__ unsigned short lds_a[64 * KPAD];      // adj tile  [64 rows][32 k] bf16 (padded)
    __shared__ unsigned short lds_b[D_FEAT * KPAD];  // node tile [256 cols][32 k] bf16 (transposed)
    __shared__ float red[256];

    const int tid  = threadIdx.x;
    const int lane = tid & 31;
    const int wave = tid >> 5;
    const int half = lane >> 4;
    const int nlo  = lane & 15;
    const int i0   = blockIdx.x * 64;

    const int mg = (wave & 1) * 2;      // m-tile pair base: tiles mg, mg+1
    const int cq = (wave >> 1) * 4;     // n-tile quarter base: tiles cq..cq+3

    const int ar  = tid >> 2;           // adj-fill row 0..63
    const int acs = (tid & 3) * 8;      // adj-fill col segment

    const int di  = tid & 63;           // diag: local row i
    const int djs = (tid >> 6) * 8;     // diag: j sub-range start

    v8f acc[2][4] = {};
    float pdiag = 0.0f;

    for (int k0 = 0; k0 < N_NODES; k0 += KTILE) {
        // ---- adj tile: f32 -> bf16 -> LDS ----
        {
            const float* src = adj + (size_t)(i0 + ar) * N_NODES + k0 + acs;
            f32x4 a0 = *reinterpret_cast<const f32x4*>(src);
            f32x4 a1 = *reinterpret_cast<const f32x4*>(src + 4);
            u32x4 p;
            p.x = pack2(a0.x, a0.y); p.y = pack2(a0.z, a0.w);
            p.z = pack2(a1.x, a1.y); p.w = pack2(a1.z, a1.w);
            *reinterpret_cast<u32x4*>(&lds_a[ar * KPAD + acs]) = p;
            if (k0 + KTILE < N_NODES)
                __builtin_prefetch(src + KTILE, 0, 3);   // global_prefetch_b8
        }
        // ---- node tile transposed: lds_b[d][k] = bf16(node[k0+k][d]); coalesced per k ----
        {
            unsigned int* dstrow = reinterpret_cast<unsigned int*>(&lds_b[tid * KPAD]);
            #pragma unroll
            for (int k = 0; k < KTILE; k += 2) {
                float lo = node[(size_t)(k0 + k)     * D_FEAT + tid];
                float hi = node[(size_t)(k0 + k + 1) * D_FEAT + tid];
                dstrow[k >> 1] = pack2(lo, hi);
            }
        }
        __syncthreads();

        // ---- diag partial: reuse adj tile in LDS; edge rows coalesced ----
        #pragma unroll
        for (int jj = 0; jj < 8; ++jj) {
            int jl = djs + jj;
            float e = edge[(size_t)(k0 + jl) * N_NODES + (i0 + di)];
            pdiag += bf2f(lds_a[di * KPAD + jl]) * e;
        }
        if (k0 + KTILE < N_NODES)
            __builtin_prefetch(&edge[(size_t)(k0 + KTILE + djs) * N_NODES + (i0 + di)], 0, 3);

        // ---- WMMA: 2 A frags, 4 B frags, each B reused for 2 WMMAs ----
        Frag a0, a1;
        load_a(a0, lds_a, (mg + 0) * 16, nlo, half);
        load_a(a1, lds_a, (mg + 1) * 16, nlo, half);
        #pragma unroll
        for (int nt = 0; nt < 4; ++nt) {
            Frag b;
            load_b(b, lds_b, (cq + nt) * 16, nlo, half);
            acc[0][nt] = __builtin_amdgcn_wmma_f32_16x16x32_bf16(
                false, a0.v, false, b.v, (short)0, acc[0][nt], false, false);
            acc[1][nt] = __builtin_amdgcn_wmma_f32_16x16x32_bf16(
                false, a1.v, false, b.v, (short)0, acc[1][nt], false, false);
        }
        __syncthreads();
    }

    // ---- store agg: VGPR r -> row r + 8*half, col = tile*16 + nlo ----
    #pragma unroll
    for (int mi = 0; mi < 2; ++mi) {
        const int mrow = (mg + mi) * 16;
        #pragma unroll
        for (int nt = 0; nt < 4; ++nt) {
            int col = (cq + nt) * 16 + nlo;
            #pragma unroll
            for (int r = 0; r < 8; ++r) {
                int row = i0 + mrow + r + 8 * half;
                agg[(size_t)row * OUT_DIM + col] = acc[mi][nt][r];
            }
        }
    }

    // ---- reduce 4 partials per i, store diag (each i owned by one block) ----
    red[tid] = pdiag;
    __syncthreads();
    if (tid < 64)
        diag[i0 + tid] = red[tid] + red[tid + 64] + red[tid + 128] + red[tid + 192];
}

// ---------------------------------------------------------------------------
// Kernel 2: out = [node | agg] @ W[0:512,:] + diag (x) W[512,:] + bias
// Same tiling; K = 512 in chunks of 32.
// ---------------------------------------------------------------------------
__global__ __launch_bounds__(256) void out_kernel(
    const float* __restrict__ node, const float* __restrict__ agg,
    const float* __restrict__ diag, const float* __restrict__ weight,
    const float* __restrict__ bias, float* __restrict__ out)
{
    __shared__ unsigned short lds_a[64 * KPAD];
    __shared__ unsigned short lds_b[OUT_DIM * KPAD];
    __shared__ float lds_diag[64];
    __shared__ float lds_w2[OUT_DIM];
    __shared__ float lds_bias[OUT_DIM];

    const int tid  = threadIdx.x;
    const int lane = tid & 31;
    const int wave = tid >> 5;
    const int half = lane >> 4;
    const int nlo  = lane & 15;
    const int i0   = blockIdx.x * 64;
    const int mg   = (wave & 1) * 2;
    const int cq   = (wave >> 1) * 4;
    const int ar   = tid >> 2;
    const int acs  = (tid & 3) * 8;

    if (tid < 64) lds_diag[tid] = diag[i0 + tid];
    lds_w2[tid]   = weight[(size_t)(2 * D_FEAT) * OUT_DIM + tid];  // last weight row
    lds_bias[tid] = bias[tid];

    v8f acc[2][4] = {};

    for (int k0 = 0; k0 < 2 * D_FEAT; k0 += KTILE) {
        // ---- A tile from node (k<256) or agg (k>=256), f32 -> bf16 ----
        {
            const float* src = (k0 < D_FEAT)
                ? node + (size_t)(i0 + ar) * D_FEAT  + k0 + acs
                : agg  + (size_t)(i0 + ar) * OUT_DIM + (k0 - D_FEAT) + acs;
            f32x4 a0 = *reinterpret_cast<const f32x4*>(src);
            f32x4 a1 = *reinterpret_cast<const f32x4*>(src + 4);
            u32x4 p;
            p.x = pack2(a0.x, a0.y); p.y = pack2(a0.z, a0.w);
            p.z = pack2(a1.x, a1.y); p.w = pack2(a1.z, a1.w);
            *reinterpret_cast<u32x4*>(&lds_a[ar * KPAD + acs]) = p;
        }
        // ---- weight tile transposed (L2-hot): lds_b[n][k] = bf16(weight[k0+k][n]) ----
        {
            unsigned int* dstrow = reinterpret_cast<unsigned int*>(&lds_b[tid * KPAD]);
            #pragma unroll
            for (int k = 0; k < KTILE; k += 2) {
                float lo = weight[(size_t)(k0 + k)     * OUT_DIM + tid];
                float hi = weight[(size_t)(k0 + k + 1) * OUT_DIM + tid];
                dstrow[k >> 1] = pack2(lo, hi);
            }
        }
        __syncthreads();

        Frag a0, a1;
        load_a(a0, lds_a, (mg + 0) * 16, nlo, half);
        load_a(a1, lds_a, (mg + 1) * 16, nlo, half);
        #pragma unroll
        for (int nt = 0; nt < 4; ++nt) {
            Frag b;
            load_b(b, lds_b, (cq + nt) * 16, nlo, half);
            acc[0][nt] = __builtin_amdgcn_wmma_f32_16x16x32_bf16(
                false, a0.v, false, b.v, (short)0, acc[0][nt], false, false);
            acc[1][nt] = __builtin_amdgcn_wmma_f32_16x16x32_bf16(
                false, a1.v, false, b.v, (short)0, acc[1][nt], false, false);
        }
        __syncthreads();
    }

    // ---- epilogue: + diag[m]*w2[n] + bias[n] ----
    #pragma unroll
    for (int mi = 0; mi < 2; ++mi) {
        const int mrow = (mg + mi) * 16;
        #pragma unroll
        for (int nt = 0; nt < 4; ++nt) {
            int col = (cq + nt) * 16 + nlo;
            float w2 = lds_w2[col];
            float bv = lds_bias[col];
            #pragma unroll
            for (int r = 0; r < 8; ++r) {
                int ml = mrow + r + 8 * half;
                out[(size_t)(i0 + ml) * OUT_DIM + col] = acc[mi][nt][r] + lds_diag[ml] * w2 + bv;
            }
        }
    }
}

extern "C" void kernel_launch(void* const* d_in, const int* in_sizes, int n_in,
                              void* d_out, int out_size, void* d_ws, size_t ws_size,
                              hipStream_t stream) {
    const float* node   = (const float*)d_in[0];   // [8192, 256]
    const float* edge   = (const float*)d_in[1];   // [8192, 8192]
    const float* adj    = (const float*)d_in[2];   // [8192, 8192]
    const float* weight = (const float*)d_in[3];   // [513, 256]
    const float* bias   = (const float*)d_in[4];   // [256]
    float* out = (float*)d_out;                    // [8192, 256]

    float* agg  = (float*)d_ws;                          // 8192*256 f32 = 8 MB
    float* diag = agg + (size_t)N_NODES * OUT_DIM;       // 8192 f32

    agg_diag_kernel<<<N_NODES / 64, 256, 0, stream>>>(adj, node, edge, agg, diag);
    out_kernel<<<N_NODES / 64, 256, 0, stream>>>(node, agg, diag, weight, bias, out);
}
